// VectorQuantizer_1168231104699
// MI455X (gfx1250) — matrix operand order repaired and checked by reading, not compiled, not run on hardware
//
#include <hip/hip_runtime.h>
#include <math.h>

// ---------------------------------------------------------------------------
// VQ-VAE vector quantizer for MI455X (gfx1250, wave32, WMMA).
// dist GEMM done as split-bf16 (hi/lo) with v_wmma_f32_16x16x32_bf16:
//   G = x . w^T  ~=  xh.wh + xh.wl + xl.wh   (covers ~16 mantissa bits)
// argmin over k of (|e_k|^2 - 2 G[n,k])  (|x|^2 per-row constant dropped)
// Main kernel: 64-row M-tile per block (4 row-tiles x 8 waves x 128 k-cols);
// codebook fragments are register-resident and feed 4 WMMA triples each.
// The d-chunk loop is kept as a runtime loop (#pragma unroll 1) so the
// A-fragments are re-read from LDS per k-tile instead of being hoisted into
// ~512 VGPRs and spilled to scratch (seen in the previous build).
// ---------------------------------------------------------------------------

typedef __bf16 bf16_t;
typedef __attribute__((ext_vector_type(16))) __bf16 v16bf;
typedef __attribute__((ext_vector_type(8)))  float  v8f;

union FragBF { uint4 u[2]; v16bf v; };

#define N_TOT   32768      // 32 * 32 * 32 flattened vectors
#define D_DIM   256
#define K_CODES 1024
#define HW      1024       // 32*32 spatial per batch
#define BETA    0.25f
#define BM      64         // rows per block in argmin kernel

// ---------------------------------------------------------------------------
// zero init of counts + loss accumulator (workspace is not re-poisoned)
// ---------------------------------------------------------------------------
__global__ void vq_zero_kernel(unsigned* __restrict__ counts,
                               double* __restrict__ loss_acc) {
  int t = blockIdx.x * 256 + threadIdx.x;
  if (t < K_CODES) counts[t] = 0u;
  if (t == 0) *loss_acc = 0.0;
}

// ---------------------------------------------------------------------------
// weight prep: bf16 hi/lo planes (row major [K,D]) + |e_k|^2
// ---------------------------------------------------------------------------
__global__ void vq_wprep_kernel(const float* __restrict__ w,
                                bf16_t* __restrict__ wh, bf16_t* __restrict__ wl,
                                float* __restrict__ wnorm) {
  __shared__ float red[256];
  const int k = blockIdx.x;
  const int d = threadIdx.x;          // D_DIM == blockDim.x == 256
  float x = w[k * D_DIM + d];
  bf16_t h = (bf16_t)x;
  bf16_t l = (bf16_t)(x - (float)h);
  wh[k * D_DIM + d] = h;
  wl[k * D_DIM + d] = l;
  red[d] = x * x;
  __syncthreads();
  for (int s = 128; s > 0; s >>= 1) {
    if (d < s) red[d] += red[d + s];
    __syncthreads();
  }
  if (d == 0) wnorm[k] = red[0];
}

// ---------------------------------------------------------------------------
// latents prep: NCHW fp32 -> [N,D] bf16 hi/lo planes, transposed via LDS
// grid: (HW/32, D/32, B)
// ---------------------------------------------------------------------------
__global__ void vq_xprep_kernel(const float* __restrict__ lat,
                                bf16_t* __restrict__ xh, bf16_t* __restrict__ xl) {
  __shared__ bf16_t th[32][33];
  __shared__ bf16_t tl[32][33];
  const int b   = blockIdx.z;
  const int d0  = blockIdx.y * 32;
  const int hw0 = blockIdx.x * 32;
  for (int i = threadIdx.x; i < 1024; i += 256) {
    int dl = i >> 5, hwl = i & 31;
    float x = lat[(size_t)b * (D_DIM * HW) + (size_t)(d0 + dl) * HW + hw0 + hwl];
    bf16_t h = (bf16_t)x;
    th[dl][hwl] = h;
    tl[dl][hwl] = (bf16_t)(x - (float)h);
  }
  __syncthreads();
  for (int i = threadIdx.x; i < 1024; i += 256) {
    int hwl = i >> 5, dl = i & 31;
    size_t o = (size_t)(b * HW + hw0 + hwl) * D_DIM + d0 + dl;
    xh[o] = th[dl][hwl];
    xl[o] = tl[dl][hwl];
  }
}

// ---------------------------------------------------------------------------
// main kernel: per block BM=64 rows x all K columns; 8 waves, 128 cols each.
// Loop order: ktile -> d-chunk (runtime loop) -> row-tile (unrolled x4).
// grid: N/64 blocks, 256 threads.
// ---------------------------------------------------------------------------
__global__ void __launch_bounds__(256)
vq_argmin_kernel(const bf16_t* __restrict__ xh, const bf16_t* __restrict__ xl,
                 const bf16_t* __restrict__ wh, const bf16_t* __restrict__ wl,
                 const float* __restrict__ wnorm,
                 int* __restrict__ idx_out, float* __restrict__ minind_out,
                 unsigned* __restrict__ counts) {
  __shared__ __align__(16) bf16_t sAh[BM * D_DIM];   // 32 KB
  __shared__ __align__(16) bf16_t sAl[BM * D_DIM];   // 32 KB
  __shared__ float sWnorm[K_CODES];                  // 4 KB
  __shared__ float sMinV[8][BM];                     // 2 KB
  __shared__ int   sMinI[8][BM];                     // 2 KB

  const int tid  = threadIdx.x;
  const int lane = tid & 31;
  const int wave = tid >> 5;                // 0..7
  const int n0   = blockIdx.x * BM;

  // stage A tile (BM rows x 256 d, hi+lo) into LDS, coalesced b128
  {
    const uint4* gsh = (const uint4*)(xh + (size_t)n0 * D_DIM);
    const uint4* gsl = (const uint4*)(xl + (size_t)n0 * D_DIM);
    uint4* dh = (uint4*)sAh;
    uint4* dl = (uint4*)sAl;
    for (int u = tid; u < (BM * D_DIM / 8); u += 256) { dh[u] = gsh[u]; dl[u] = gsl[u]; }
    for (int k = tid; k < K_CODES; k += 256) sWnorm[k] = wnorm[k];
  }
  __syncthreads();

  const int hf  = lane >> 4;                // 0/1 half of wave
  const int l16 = lane & 15;

  float runv[4][8];
  int   runi[4][8];
#pragma unroll
  for (int rt = 0; rt < 4; ++rt)
#pragma unroll
    for (int v = 0; v < 8; ++v) { runv[rt][v] = 3.4e38f; runi[rt][v] = 0; }

#pragma unroll 1
  for (int t = 0; t < 8; ++t) {
    const int kcol = wave * 128 + t * 16 + l16;  // B column this lane supplies
    const bf16_t* bRowH = wh + (size_t)kcol * D_DIM;
    const bf16_t* bRowL = wl + (size_t)kcol * D_DIM;

    v8f acc[4];
#pragma unroll
    for (int rt = 0; rt < 4; ++rt) acc[rt] = (v8f){0.f,0.f,0.f,0.f,0.f,0.f,0.f,0.f};

    // runtime loop: keeps A-fragment LDS loads inside (prevents 512-VGPR
    // hoist + scratch spilling); fragments are re-read from LDS per k-tile.
#pragma unroll 1
    for (int ch = 0; ch < 8; ++ch) {
      const int d0 = ch * 32;
      FragBF bh, bl;
      // B 32x16 bf16: elems 0..7 -> d = d0 + hf*8 + 0..7, elems 8..15 -> d = d0 + 16 + hf*8 + 0..7
      bh.u[0] = *(const uint4*)(bRowH + d0 + hf * 8);
      bh.u[1] = *(const uint4*)(bRowH + d0 + 16 + hf * 8);
      bl.u[0] = *(const uint4*)(bRowL + d0 + hf * 8);
      bl.u[1] = *(const uint4*)(bRowL + d0 + 16 + hf * 8);

#pragma unroll
      for (int rt = 0; rt < 4; ++rt) {
        const bf16_t* aRowH = sAh + (rt * 16 + l16) * D_DIM;
        const bf16_t* aRowL = sAl + (rt * 16 + l16) * D_DIM;
        FragBF ah, al;
        // A 16x32 bf16: elems 0..7  -> d = d0 + hf*8 + 0..7
        //               elems 8..15 -> d = d0 + 16 + hf*8 + 0..7
        ah.u[0] = *(const uint4*)(aRowH + d0 + hf * 8);
        ah.u[1] = *(const uint4*)(aRowH + d0 + 16 + hf * 8);
        al.u[0] = *(const uint4*)(aRowL + d0 + hf * 8);
        al.u[1] = *(const uint4*)(aRowL + d0 + 16 + hf * 8);

        acc[rt] = __builtin_amdgcn_wmma_f32_16x16x32_bf16(false, ah.v, false, bh.v,
                                                          (short)0, acc[rt], false, false);
        acc[rt] = __builtin_amdgcn_wmma_f32_16x16x32_bf16(false, ah.v, false, bl.v,
                                                          (short)0, acc[rt], false, false);
        acc[rt] = __builtin_amdgcn_wmma_f32_16x16x32_bf16(false, al.v, false, bh.v,
                                                          (short)0, acc[rt], false, false);
      }
    }

    // C layout: VGPR v, lanes 0-15 -> row v, col lane; lanes 16-31 -> row v+8
    const float wn = sWnorm[kcol];
#pragma unroll
    for (int rt = 0; rt < 4; ++rt) {
#pragma unroll
      for (int v = 0; v < 8; ++v) {
        float dist = wn - 2.0f * acc[rt][v];
        if (dist < runv[rt][v]) { runv[rt][v] = dist; runi[rt][v] = kcol; }
      }
    }
  }

  // reduce across the 16 column-lanes of each half (rows differ per half)
#pragma unroll
  for (int m = 1; m < 16; m <<= 1) {
#pragma unroll
    for (int rt = 0; rt < 4; ++rt) {
#pragma unroll
      for (int v = 0; v < 8; ++v) {
        float ov = __shfl_xor(runv[rt][v], m, 32);
        int   oi = __shfl_xor(runi[rt][v], m, 32);
        if (ov < runv[rt][v] || (ov == runv[rt][v] && oi < runi[rt][v])) {
          runv[rt][v] = ov; runi[rt][v] = oi;
        }
      }
    }
  }
  if (l16 == 0) {
#pragma unroll
    for (int rt = 0; rt < 4; ++rt) {
#pragma unroll
      for (int v = 0; v < 8; ++v) {
        sMinV[wave][rt * 16 + hf * 8 + v] = runv[rt][v];
        sMinI[wave][rt * 16 + hf * 8 + v] = runi[rt][v];
      }
    }
  }
  __syncthreads();

  // final reduce across 8 waves (increasing wave = increasing k range)
  if (tid < BM) {
    float bv = sMinV[0][tid];
    int   bi = sMinI[0][tid];
#pragma unroll
    for (int w2 = 1; w2 < 8; ++w2) {
      float ov = sMinV[w2][tid];
      int   oi = sMinI[w2][tid];
      if (ov < bv || (ov == bv && oi < bi)) { bv = ov; bi = oi; }
    }
    idx_out[n0 + tid]    = bi;
    minind_out[n0 + tid] = (float)bi;
    atomicAdd(&counts[bi], 1u);
  }
}

// ---------------------------------------------------------------------------
// z_q gather + NCHW write (transposed via LDS) fused with loss reduction
// grid: (HW/32, D/32, B)
// ---------------------------------------------------------------------------
__global__ void vq_zq_loss_kernel(const float* __restrict__ lat,
                                  const float* __restrict__ w,
                                  const int* __restrict__ idx,
                                  float* __restrict__ zq_out,
                                  double* __restrict__ loss_acc) {
  __shared__ float tile[32][33];
  __shared__ int sidx[32];
  __shared__ double red[256];
  const int b   = blockIdx.z;
  const int d0  = blockIdx.y * 32;
  const int hw0 = blockIdx.x * 32;
  const int n0  = b * HW + hw0;

  if (threadIdx.x < 32) sidx[threadIdx.x] = idx[n0 + threadIdx.x];
  __syncthreads();

  // gather codebook rows (coalesced along d)
  for (int i = threadIdx.x; i < 1024; i += 256) {
    int nl = i >> 5, dl = i & 31;
    tile[nl][dl] = w[(size_t)sidx[nl] * D_DIM + d0 + dl];
  }
  __syncthreads();

  // write NCHW (coalesced along hw) + accumulate (z_q - x)^2
  double acc = 0.0;
  for (int i = threadIdx.x; i < 1024; i += 256) {
    int dl = i >> 5, nl = i & 31;
    size_t o = (size_t)b * (D_DIM * HW) + (size_t)(d0 + dl) * HW + hw0 + nl;
    float zq = tile[nl][dl];
    float x  = lat[o];
    zq_out[o] = zq;
    float df = zq - x;
    acc += (double)(df * df);
  }
  red[threadIdx.x] = acc;
  __syncthreads();
  for (int s = 128; s > 0; s >>= 1) {
    if (threadIdx.x < s) red[threadIdx.x] += red[threadIdx.x + s];
    __syncthreads();
  }
  if (threadIdx.x == 0) atomicAdd(loss_acc, red[0]);
}

// ---------------------------------------------------------------------------
// one-hot [N, K] output (pure bandwidth, b128 stores)
// ---------------------------------------------------------------------------
__global__ void vq_onehot_kernel(const int* __restrict__ idx,
                                 float4* __restrict__ me4) {
  size_t g = (size_t)blockIdx.x * 256 + threadIdx.x;   // float4 index
  int n  = (int)(g >> 8);                              // 256 float4 per row
  int k0 = (int)((g & 255) << 2);
  int bi = idx[n];
  float4 v;
  v.x = (k0 + 0 == bi) ? 1.0f : 0.0f;
  v.y = (k0 + 1 == bi) ? 1.0f : 0.0f;
  v.z = (k0 + 2 == bi) ? 1.0f : 0.0f;
  v.w = (k0 + 3 == bi) ? 1.0f : 0.0f;
  me4[g] = v;
}

// ---------------------------------------------------------------------------
// scalars: loss & perplexity
// ---------------------------------------------------------------------------
__global__ void vq_final_kernel(const unsigned* __restrict__ counts,
                                const double* __restrict__ loss_acc,
                                float* __restrict__ loss_out,
                                float* __restrict__ perp_out) {
  __shared__ float red[256];
  float acc = 0.f;
  for (int k = threadIdx.x; k < K_CODES; k += 256) {
    float em = (float)counts[k] * (1.0f / (float)N_TOT);
    acc += em * logf(em + 1e-10f);
  }
  red[threadIdx.x] = acc;
  __syncthreads();
  for (int s = 128; s > 0; s >>= 1) {
    if (threadIdx.x < s) red[threadIdx.x] += red[threadIdx.x + s];
    __syncthreads();
  }
  if (threadIdx.x == 0) {
    *perp_out = expf(-red[0]);
    // loss = (1 + beta) * mean((z_q - x)^2) over N*D elements
    *loss_out = (float)(*loss_acc * ((1.0 + (double)BETA) / (double)(N_TOT * D_DIM)));
  }
}

// ---------------------------------------------------------------------------
extern "C" void kernel_launch(void* const* d_in, const int* in_sizes, int n_in,
                              void* d_out, int out_size, void* d_ws, size_t ws_size,
                              hipStream_t stream) {
  (void)in_sizes; (void)n_in; (void)out_size; (void)ws_size;
  const float* lat = (const float*)d_in[0];   // [32,256,32,32]
  const float* w   = (const float*)d_in[1];   // [1024,256]
  float* out = (float*)d_out;
  char*  ws  = (char*)d_ws;

  // workspace layout (bytes)
  bf16_t*   wh       = (bf16_t*)(ws + 0);               // 512 KB
  bf16_t*   wl       = (bf16_t*)(ws + 524288);          // 512 KB
  bf16_t*   xh       = (bf16_t*)(ws + 1048576);         // 16 MB
  bf16_t*   xl       = (bf16_t*)(ws + 17825792);        // 16 MB
  float*    wnorm    = (float*)(ws + 34603008);         // 4 KB
  int*      idx      = (int*)(ws + 34607104);           // 128 KB
  unsigned* counts   = (unsigned*)(ws + 34738176);      // 4 KB
  double*   loss_acc = (double*)(ws + 34742272);        // 8 B

  // output layout: loss(1) | z_q NCHW(8388608) | perplexity(1) |
  //                min_embed(33554432) | min_ind(32768)
  float* loss_out = out + 0;
  float* zq_out   = out + 1;
  float* perp_out = out + 8388609;
  float* me_out   = out + 8388610;
  float* mi_out   = out + 41943042;

  dim3 tiles(HW / 32, D_DIM / 32, 32);

  vq_zero_kernel<<<4, 256, 0, stream>>>(counts, loss_acc);
  vq_wprep_kernel<<<K_CODES, 256, 0, stream>>>(w, wh, wl, wnorm);
  vq_xprep_kernel<<<tiles, 256, 0, stream>>>(lat, xh, xl);
  vq_argmin_kernel<<<N_TOT / BM, 256, 0, stream>>>(xh, xl, wh, wl, wnorm,
                                                   idx, mi_out, counts);
  vq_zq_loss_kernel<<<tiles, 256, 0, stream>>>(lat, w, idx, zq_out, loss_acc);
  vq_onehot_kernel<<<(N_TOT * K_CODES) / 1024, 256, 0, stream>>>(
      idx, (float4*)me_out);
  vq_final_kernel<<<1, 256, 0, stream>>>(counts, loss_acc, loss_out, perp_out);
}
